// EinFFT_17377437680461
// MI455X (gfx1250) — matrix operand order repaired and checked by reading, not compiled, not run on hardware
//
#include <hip/hip_runtime.h>
#include <hip/hip_bf16.h>

typedef __attribute__((ext_vector_type(16))) __bf16 v16bf;
typedef __attribute__((ext_vector_type(8)))  float  v8f;

#define TOKENS   32768      // B*N = 8*4096
#define NFFT     4096
#define CCH      512
#define BS       128
#define NBLK     4
#define LAM      0.01f
#define PI_F     3.14159265358979f

#define FRAG_ELEMS 512                    // 32x16 bf16 tile, 16 per lane
#define MAT_FRAGS  (8 * 4)                // nt * kk fragments per 128x128 matrix
#define MAT_ELEMS  (MAT_FRAGS * FRAG_ELEMS)   // 16384 = 128*128
// packed layout: ((((mat*4+blk)*3 + var)*8 + nt)*4 + kk)*512 + lane*16 + e
// var: 0 = +W_real, 1 = +W_imag, 2 = -W_imag

// ---------------------------------------------------------------------------
// WMMA helpers
// ---------------------------------------------------------------------------
__device__ __forceinline__ v8f wmma_bf16(v16bf a, v16bf b, v8f c) {
  return __builtin_amdgcn_wmma_f32_16x16x32_bf16(false, a, false, b, (short)0, c,
                                                 false, false);
}

__device__ __forceinline__ v8f bias_bcast(float bv) {
  v8f c;
#pragma unroll
  for (int j = 0; j < 8; ++j) c[j] = bv;
  return c;
}

// A fragment (16x32 bf16): lane L -> row m=L&15, hi=L>>4.
// element e<8  : k = kk*32 + hi*8 + e
// element e>=8 : k = kk*32 + 16 + hi*8 + (e-8)
__device__ __forceinline__ v16bf load_a_frag(const float* __restrict__ row, int kk, int hi) {
  const int k0 = kk * 32 + hi * 8;
  const float4 u0 = *(const float4*)(row + k0);
  const float4 u1 = *(const float4*)(row + k0 + 4);
  const float4 u2 = *(const float4*)(row + k0 + 16);
  const float4 u3 = *(const float4*)(row + k0 + 20);
  v16bf f;
  f[0]  = (__bf16)u0.x; f[1]  = (__bf16)u0.y; f[2]  = (__bf16)u0.z; f[3]  = (__bf16)u0.w;
  f[4]  = (__bf16)u1.x; f[5]  = (__bf16)u1.y; f[6]  = (__bf16)u1.z; f[7]  = (__bf16)u1.w;
  f[8]  = (__bf16)u2.x; f[9]  = (__bf16)u2.y; f[10] = (__bf16)u2.z; f[11] = (__bf16)u2.w;
  f[12] = (__bf16)u3.x; f[13] = (__bf16)u3.y; f[14] = (__bf16)u3.z; f[15] = (__bf16)u3.w;
  return f;
}

__device__ __forceinline__ v16bf load_a_frag_lds(const __bf16* row, int kk, int hi) {
  const int k0 = kk * 32 + hi * 8;
  v16bf f;
#pragma unroll
  for (int e = 0; e < 8; ++e) f[e] = row[k0 + e];
#pragma unroll
  for (int e = 0; e < 8; ++e) f[8 + e] = row[k0 + 16 + e];
  return f;
}

// ---------------------------------------------------------------------------
// K0: one-shot weight pack: f32 (2,4,128,128) x2 -> fragment-major bf16 with
// +Wr / +Wi / -Wi variants. Total 0.75 MB, stays L2-resident thereafter.
// ---------------------------------------------------------------------------
__global__ __launch_bounds__(256) void k_pack_w(const float* __restrict__ w1,
                                                const float* __restrict__ w2,
                                                __bf16* __restrict__ wp) {
  const int gid = blockIdx.x * 256 + threadIdx.x;     // < 393216
  const int e    = gid & 15;
  const int lane = (gid >> 4) & 31;
  const int kk   = (gid >> 9) & 3;
  const int nt   = (gid >> 11) & 7;
  const int rest = gid >> 14;                         // 0..23
  const int var  = rest % 3;
  const int blkm = rest / 3;                          // 0..7
  const int blk  = blkm & 3;
  const int mat  = blkm >> 2;

  const int ri  = (var == 0) ? 0 : 1;
  const float sg = (var == 2) ? -1.f : 1.f;
  const int hi  = lane >> 4;
  const int d   = kk * 32 + hi * 8 + (e & 7) + ((e >> 3) << 4); // input dim (K)
  const int kc  = nt * 16 + (lane & 15);              // output dim (N)
  const float* w = mat ? w2 : w1;
  const float v = w[((size_t)(ri * NBLK + blk) * BS + d) * BS + kc];
  wp[gid] = (__bf16)(sg * v);
}

// ---------------------------------------------------------------------------
// K1: forward length-4096 FFT along tokens (radix-2 DIF, bit-reversed output)
// ---------------------------------------------------------------------------
__global__ __launch_bounds__(256) void k_fft_fwd(const float* __restrict__ x,
                                                 float* __restrict__ wre,
                                                 float* __restrict__ wim) {
  __shared__ float sr[NFFT];
  __shared__ float si[NFFT];
  const int b = blockIdx.x >> 9;
  const int c = blockIdx.x & 511;
  const float* xp = x + (size_t)b * NFFT * CCH + c;
  for (int n = threadIdx.x; n < NFFT; n += 256) {
    sr[n] = xp[(size_t)n * CCH];
    si[n] = 0.f;
  }
  __syncthreads();
#pragma unroll 1
  for (int s = 0; s < 12; ++s) {
    const int half = 2048 >> s;
    for (int j = threadIdx.x; j < 2048; j += 256) {
      const int pos = j & (half - 1);
      const int i0  = ((j >> (11 - s)) << (12 - s)) + pos;
      const int i1  = i0 + half;
      const float ang = -PI_F * (float)pos / (float)half;
      float wi, wr;
      __sincosf(ang, &wi, &wr);
      const float ar = sr[i0], ai = si[i0];
      const float br = sr[i1], bi = si[i1];
      const float dr = ar - br, di = ai - bi;
      sr[i0] = ar + br;
      si[i0] = ai + bi;
      sr[i1] = dr * wr - di * wi;
      si[i1] = dr * wi + di * wr;
    }
    __syncthreads();
  }
  float* orp = wre + (size_t)b * NFFT * CCH + c;
  float* oip = wim + (size_t)b * NFFT * CCH + c;
  for (int n = threadIdx.x; n < NFFT; n += 256) {
    orp[(size_t)n * CCH] = sr[n] * 0.015625f;
    oip[(size_t)n * CCH] = si[n] * 0.015625f;
  }
}

// ---------------------------------------------------------------------------
// K2a/K2c: length-4 FFT across the block axis, ortho 1/2
// ---------------------------------------------------------------------------
__global__ __launch_bounds__(256) void k_blk_fft(float* __restrict__ wre,
                                                 float* __restrict__ wim,
                                                 int inverse) {
  const int u = blockIdx.x * 256 + threadIdx.x;  // < TOKENS*BS
  const int t = u >> 7;
  const int d = u & 127;
  const size_t base = (size_t)t * CCH + d;
  const float sg = inverse ? -1.f : 1.f;
  const float r0 = wre[base], r1 = wre[base + 128], r2 = wre[base + 256], r3 = wre[base + 384];
  const float i0 = wim[base], i1 = wim[base + 128], i2 = wim[base + 256], i3 = wim[base + 384];
  const float y0r = r0 + r1 + r2 + r3,            y0i = i0 + i1 + i2 + i3;
  const float y2r = r0 - r1 + r2 - r3,            y2i = i0 - i1 + i2 - i3;
  const float y1r = r0 + sg * i1 - r2 - sg * i3,  y1i = i0 - sg * r1 - i2 + sg * r3;
  const float y3r = r0 - sg * i1 - r2 + sg * i3,  y3i = i0 + sg * r1 - i2 - sg * r3;
  wre[base]       = 0.5f * y0r;  wim[base]       = 0.5f * y0i;
  wre[base + 128] = 0.5f * y1r;  wim[base + 128] = 0.5f * y1i;
  wre[base + 256] = 0.5f * y2r;  wim[base + 256] = 0.5f * y2i;
  wre[base + 384] = 0.5f * y3r;  wim[base + 384] = 0.5f * y3i;
}

// ---------------------------------------------------------------------------
// K2b: complex block-diagonal MLP + softshrink, in place on (wre, wim).
// grid = (TOKENS/128, NBLK), 256 threads = 8 waves; each wave: 16 tokens x 128.
// B matrices come from the pre-packed fragment-major bf16 buffer: one lane =
// 32 contiguous bytes -> two global_load_b128, no cvt in the hot loop.
// ---------------------------------------------------------------------------
__global__ __launch_bounds__(256) void k_mlp(float* __restrict__ wre,
                                             float* __restrict__ wim,
                                             const __bf16* __restrict__ wp,
                                             const float* __restrict__ b1,
                                             const float* __restrict__ b2) {
  __shared__ __bf16 lr[8][16][BS];
  __shared__ __bf16 li[8][16][BS];
  const int wave = threadIdx.x >> 5;
  const int lane = threadIdx.x & 31;
  const int m    = lane & 15;
  const int hi   = lane >> 4;
  const int blk  = blockIdx.y;
  const int tok0 = (blockIdx.x * 8 + wave) * 16;

  const float* rrow = wre + (size_t)(tok0 + m) * CCH + blk * BS;
  const float* irow = wim + (size_t)(tok0 + m) * CCH + blk * BS;

  v16bf ar[4], ai[4];
#pragma unroll
  for (int kk = 0; kk < 4; ++kk) {
    ar[kk] = load_a_frag(rrow, kk, hi);
    ai[kk] = load_a_frag(irow, kk, hi);
  }

  // ---- layer 1 ----
  const __bf16* p1  = wp + (size_t)(0 * NBLK + blk) * 3 * MAT_ELEMS + lane * 16;
  const float*  b1r = b1 + (0 * NBLK + blk) * BS;
  const float*  b1i = b1 + (1 * NBLK + blk) * BS;
#pragma unroll 1
  for (int nt = 0; nt < 8; ++nt) {
    v8f cr = bias_bcast(b1r[nt * 16 + m]);
    v8f ci = bias_bcast(b1i[nt * 16 + m]);
    const __bf16* pf = p1 + (size_t)(nt * 4) * FRAG_ELEMS;
#pragma unroll
    for (int kk = 0; kk < 4; ++kk) {
      const v16bf br  = *(const v16bf*)(pf + 0 * MAT_ELEMS + kk * FRAG_ELEMS);
      const v16bf bi  = *(const v16bf*)(pf + 1 * MAT_ELEMS + kk * FRAG_ELEMS);
      const v16bf bin = *(const v16bf*)(pf + 2 * MAT_ELEMS + kk * FRAG_ELEMS);
      cr = wmma_bf16(ar[kk], br, cr);
      cr = wmma_bf16(ai[kk], bin, cr);
      ci = wmma_bf16(ar[kk], bi, ci);
      ci = wmma_bf16(ai[kk], br, ci);
    }
#pragma unroll
    for (int j = 0; j < 8; ++j) {
      const int   row = j + hi * 8;
      const int   col = nt * 16 + m;
      const float vr  = cr[j] > 0.f ? cr[j] : 0.f;
      const float vi  = ci[j] > 0.f ? ci[j] : 0.f;
      lr[wave][row][col] = (__bf16)vr;
      li[wave][row][col] = (__bf16)vi;
    }
  }
  __syncthreads();

  // ---- layer 2 + softshrink, write back in place ----
#pragma unroll
  for (int kk = 0; kk < 4; ++kk) {
    ar[kk] = load_a_frag_lds(&lr[wave][m][0], kk, hi);
    ai[kk] = load_a_frag_lds(&li[wave][m][0], kk, hi);
  }
  const __bf16* p2  = wp + (size_t)(1 * NBLK + blk) * 3 * MAT_ELEMS + lane * 16;
  const float*  b2r = b2 + (0 * NBLK + blk) * BS;
  const float*  b2i = b2 + (1 * NBLK + blk) * BS;
#pragma unroll 1
  for (int nt = 0; nt < 8; ++nt) {
    v8f cr = bias_bcast(b2r[nt * 16 + m]);
    v8f ci = bias_bcast(b2i[nt * 16 + m]);
    const __bf16* pf = p2 + (size_t)(nt * 4) * FRAG_ELEMS;
#pragma unroll
    for (int kk = 0; kk < 4; ++kk) {
      const v16bf br  = *(const v16bf*)(pf + 0 * MAT_ELEMS + kk * FRAG_ELEMS);
      const v16bf bi  = *(const v16bf*)(pf + 1 * MAT_ELEMS + kk * FRAG_ELEMS);
      const v16bf bin = *(const v16bf*)(pf + 2 * MAT_ELEMS + kk * FRAG_ELEMS);
      cr = wmma_bf16(ar[kk], br, cr);
      cr = wmma_bf16(ai[kk], bin, cr);
      ci = wmma_bf16(ar[kk], bi, ci);
      ci = wmma_bf16(ai[kk], br, ci);
    }
#pragma unroll
    for (int j = 0; j < 8; ++j) {
      const int row = j + hi * 8;
      const int col = nt * 16 + m;
      float vr = cr[j];
      float vi = ci[j];
      vr = vr > LAM ? vr - LAM : (vr < -LAM ? vr + LAM : 0.f);
      vi = vi > LAM ? vi - LAM : (vi < -LAM ? vi + LAM : 0.f);
      wre[(size_t)(tok0 + row) * CCH + blk * BS + col] = vr;
      wim[(size_t)(tok0 + row) * CCH + blk * BS + col] = vi;
    }
  }
}

// ---------------------------------------------------------------------------
// K3: inverse length-4096 FFT (DIT, bit-reversed in -> natural out), real part
// ---------------------------------------------------------------------------
__global__ __launch_bounds__(256) void k_fft_inv(const float* __restrict__ wre,
                                                 const float* __restrict__ wim,
                                                 float* __restrict__ out) {
  __shared__ float sr[NFFT];
  __shared__ float si[NFFT];
  const int b = blockIdx.x >> 9;
  const int c = blockIdx.x & 511;
  const float* rp = wre + (size_t)b * NFFT * CCH + c;
  const float* ip = wim + (size_t)b * NFFT * CCH + c;
  for (int n = threadIdx.x; n < NFFT; n += 256) {
    sr[n] = rp[(size_t)n * CCH];
    si[n] = ip[(size_t)n * CCH];
  }
  __syncthreads();
#pragma unroll 1
  for (int s = 0; s < 12; ++s) {
    const int half = 1 << s;
    for (int j = threadIdx.x; j < 2048; j += 256) {
      const int pos = j & (half - 1);
      const int i0  = ((j >> s) << (s + 1)) + pos;
      const int i1  = i0 + half;
      const float ang = PI_F * (float)pos / (float)half;
      float wi, wr;
      __sincosf(ang, &wi, &wr);
      const float br = sr[i1], bi = si[i1];
      const float tr = br * wr - bi * wi;
      const float ti = br * wi + bi * wr;
      const float arv = sr[i0], aiv = si[i0];
      sr[i0] = arv + tr;
      si[i0] = aiv + ti;
      sr[i1] = arv - tr;
      si[i1] = aiv - ti;
    }
    __syncthreads();
  }
  float* op = out + (size_t)b * NFFT * CCH + c;
  for (int n = threadIdx.x; n < NFFT; n += 256)
    op[(size_t)n * CCH] = sr[n] * 0.015625f;
}

// ---------------------------------------------------------------------------
extern "C" void kernel_launch(void* const* d_in, const int* in_sizes, int n_in,
                              void* d_out, int out_size, void* d_ws, size_t ws_size,
                              hipStream_t stream) {
  (void)in_sizes; (void)n_in; (void)out_size; (void)ws_size;
  const float* x  = (const float*)d_in[0];
  const float* w1 = (const float*)d_in[1];
  const float* w2 = (const float*)d_in[2];
  const float* b1 = (const float*)d_in[3];
  const float* b2 = (const float*)d_in[4];
  float* out = (float*)d_out;

  float*  wre = (float*)d_ws;                         // 64 MB
  float*  wim = wre + (size_t)TOKENS * CCH;           // 64 MB
  __bf16* wp  = (__bf16*)(wim + (size_t)TOKENS * CCH);// 0.75 MB packed weights

  k_pack_w<<<1536, 256, 0, stream>>>(w1, w2, wp);
  k_fft_fwd<<<NFFT, 256, 0, stream>>>(x, wre, wim);
  k_blk_fft<<<(TOKENS * BS) / 256, 256, 0, stream>>>(wre, wim, 0);
  k_mlp<<<dim3(TOKENS / 128, NBLK), 256, 0, stream>>>(wre, wim, wp, b1, b2);
  k_blk_fft<<<(TOKENS * BS) / 256, 256, 0, stream>>>(wre, wim, 1);
  k_fft_inv<<<NFFT, 256, 0, stream>>>(wre, wim, out);
}
